// CausalSelfAttention_81423989997760
// MI455X (gfx1250) — matrix operand order repaired and checked by reading, compiled, not run on hardware
//
#include <hip/hip_runtime.h>
#include <stdint.h>

#define EMBED 1024
#define NHEAD 16
#define HDIM  64
#define SEQ   2048
#define BATCH 4

typedef __attribute__((ext_vector_type(16))) __bf16        v16bf;
typedef __attribute__((ext_vector_type(8)))  float         v8f;
typedef __attribute__((ext_vector_type(4)))  unsigned int  u32x4;
typedef __attribute__((ext_vector_type(4)))  float         f32x4;
typedef __attribute__((ext_vector_type(4)))  unsigned short u16x4;

typedef __attribute__((address_space(3))) unsigned short lds_u16;

union FragU { u32x4 q[2]; v16bf bf; };

__device__ __forceinline__ unsigned short f2bf(float f) {
  unsigned int u = __float_as_uint(f);
  u += 0x7FFFu + ((u >> 16) & 1u);          // round-to-nearest-even
  return (unsigned short)(u >> 16);
}
__device__ __forceinline__ u32x4 ld128(const unsigned short* p) {
  return *reinterpret_cast<const u32x4*>(p);
}
#define NEG_INF (__int_as_float(0xff800000))

// ---------------- fp32 -> bf16 conversion (vectorized x4) ----------------
__global__ void cvt_bf16(const float* __restrict__ in, unsigned short* __restrict__ out, int n4) {
  int i = blockIdx.x * 256 + threadIdx.x;
  if (i >= n4) return;
  f32x4 f = reinterpret_cast<const f32x4*>(in)[i];
  u16x4 o;
  o[0] = f2bf(f[0]); o[1] = f2bf(f[1]); o[2] = f2bf(f[2]); o[3] = f2bf(f[3]);
  reinterpret_cast<u16x4*>(out)[i] = o;
}

// ---------------- bf16 GEMM: out[m,n] = sum_k A[m,k] * W[n,k] ----------------
// Block = 8 waves (2x4), wave tile 64x64 (4x4 WMMA tiles), K stepped by 32.
// mode 0: write f32 to outF [M,N].  mode 1: qkv split -> q/k/v bf16 [B,H,T,D].
__global__ __launch_bounds__(256)
void gemm_wmma(const unsigned short* __restrict__ A,
               const unsigned short* __restrict__ W,
               int N, int mode,
               float* __restrict__ outF,
               unsigned short* __restrict__ qb,
               unsigned short* __restrict__ kb,
               unsigned short* __restrict__ vb) {
  const int K    = EMBED;
  const int lane = threadIdx.x & 31;
  const int wave = threadIdx.x >> 5;
  const int lm   = lane & 15;
  const int lh   = lane >> 4;
  const int mBase = blockIdx.x * 128 + (wave & 1) * 64;
  const int nBase = blockIdx.y * 256 + (wave >> 1) * 64;

  v8f zero = {};
  v8f acc[4][4];
#pragma unroll
  for (int i = 0; i < 4; ++i)
#pragma unroll
    for (int j = 0; j < 4; ++j) acc[i][j] = zero;

  const unsigned short* aRow = A + (size_t)(mBase + lm) * K + lh * 16;
  const unsigned short* bRow = W + (size_t)(nBase + lm) * K + lh * 16;

  for (int k0 = 0; k0 < K; k0 += 32) {
    FragU a[4], b[4];
#pragma unroll
    for (int mt = 0; mt < 4; ++mt) {
      const unsigned short* p = aRow + (size_t)mt * 16 * K + k0;
      a[mt].q[0] = ld128(p);        // k:  base..base+7
      a[mt].q[1] = ld128(p + 8);    // k:  base+8..base+15
    }
#pragma unroll
    for (int nt = 0; nt < 4; ++nt) {
      const unsigned short* p = bRow + (size_t)nt * 16 * K + k0;
      b[nt].q[0] = ld128(p);        // 16 contiguous k values
      b[nt].q[1] = ld128(p + 8);
    }
#pragma unroll
    for (int mt = 0; mt < 4; ++mt)
#pragma unroll
      for (int nt = 0; nt < 4; ++nt)
        acc[mt][nt] = __builtin_amdgcn_wmma_f32_16x16x32_bf16(
            false, a[mt].bf, false, b[nt].bf, (short)0, acc[mt][nt], false, false);
  }

#pragma unroll
  for (int mt = 0; mt < 4; ++mt)
#pragma unroll
    for (int nt = 0; nt < 4; ++nt)
#pragma unroll
      for (int i = 0; i < 8; ++i) {
        int row = mBase + mt * 16 + i + 8 * lh;
        int col = nBase + nt * 16 + lm;
        float val = acc[mt][nt][i];
        if (mode == 0) {
          outF[(size_t)row * N + col] = val;
        } else {
          int part = col >> 10;            // which of q/k/v
          int c = col & 1023;
          int h = c >> 6, d = c & 63;
          int bi = row >> 11;              // / SEQ
          int t  = row & 2047;
          size_t idx = (((size_t)bi * NHEAD + h) * SEQ + t) * HDIM + d;
          unsigned short bv = f2bf(val);
          if (part == 0) qb[idx] = bv;
          else if (part == 1) kb[idx] = bv;
          else vb[idx] = bv;
        }
      }
}

// ---------------- causal flash attention ----------------
// grid: (SEQ/128, B*H). 8 waves; wave w owns 16 q rows.
// K tiles DMA'd global->LDS with GLOBAL_LOAD_ASYNC_TO_LDS_B128 (ASYNCcnt),
// double-buffered so tile i+1 streams while tile i computes. V is transposed
// into LDS via VGPR scatter (data needed in registers for transpose anyway).
struct SMem {
  unsigned short Ks[2][32][64];   // K tile, row-major, double buffered
  unsigned short Vt[2][64][40];   // V tile transposed (padded), double buffered
  unsigned short Pb[8][16][32];   // per-wave P staging (C-layout -> A-layout)
};

__global__ __launch_bounds__(256)
void flash_attn(const unsigned short* __restrict__ Q,
                const unsigned short* __restrict__ Km,
                const unsigned short* __restrict__ Vm,
                unsigned short* __restrict__ O) {
  __shared__ SMem sm;

  const int lane = threadIdx.x & 31;
  const int wave = threadIdx.x >> 5;
  const int lm = lane & 15, lh = lane >> 4;
  const int bh     = blockIdx.y;
  const int qBlock = blockIdx.x * 128;
  const int qBase  = qBlock + wave * 16;

  const unsigned short* Qh = Q  + (size_t)bh * SEQ * HDIM;
  const unsigned short* Kh = Km + (size_t)bh * SEQ * HDIM;
  const unsigned short* Vh = Vm + (size_t)bh * SEQ * HDIM;

  FragU qf[2];
  {
    const unsigned short* p = Qh + (size_t)(qBase + lm) * HDIM + lh * 16;
    qf[0].q[0] = ld128(p);       qf[0].q[1] = ld128(p + 8);    // d 0..31
    qf[1].q[0] = ld128(p + 32);  qf[1].q[1] = ld128(p + 40);   // d 32..63
  }

  v8f zero = {};
  v8f oacc[4];
  float mstat[8], lstat[8];
#pragma unroll
  for (int i = 0; i < 4; ++i) oacc[i] = zero;
#pragma unroll
  for (int i = 0; i < 8; ++i) { mstat[i] = NEG_INF; lstat[i] = 0.0f; }

  const int tid  = threadIdx.x;
  const int kRow = tid >> 3;             // 0..31
  const int kCol = (tid & 7) * 8;        // 0..56
  const int numKT = (qBlock + 128) >> 5; // causal: only tiles <= block's max q

  // stage tile kt into buffer buf: K via async DMA, V via VGPR transpose
  auto stage = [&](int kt, int buf) {
    const int ktBase = kt * 32;
    const unsigned short* kp = Kh + (size_t)(ktBase + kRow) * HDIM + kCol;
    lds_u16* kd = (lds_u16*)&sm.Ks[buf][kRow][kCol];
    asm volatile("global_load_async_to_lds_b128 %0, %1, off"
                 :: "v"(kd), "v"(kp) : "memory");
    u32x4 vd = ld128(Vh + (size_t)(ktBase + kRow) * HDIM + kCol);
    const unsigned short* vs = reinterpret_cast<const unsigned short*>(&vd);
#pragma unroll
    for (int j = 0; j < 8; ++j) sm.Vt[buf][kCol + j][kRow] = vs[j];
  };

  stage(0, 0);   // prologue

  for (int kt = 0; kt < numKT; ++kt) {
    const int ktBase = kt * 32;
    const int cur = kt & 1;
    asm volatile("s_wait_asynccnt 0x0" ::: "memory");  // my DMA into cur done
    __syncthreads();                                   // everyone's writes visible
    if (kt + 1 < numKT) stage(kt + 1, cur ^ 1);        // overlap next tile's traffic

    if (ktBase <= qBase + 15) {          // skip fully-masked tiles (uniform per wave)
      // S = Q K^T : two 16x16 tiles (keys ktBase..+15, +16..+31)
      v8f s0 = zero, s1 = zero;
      FragU kf;
#pragma unroll
      for (int nt = 0; nt < 2; ++nt) {
        const unsigned short* p0 = &sm.Ks[cur][nt * 16 + lm][lh * 16];
        kf.q[0] = ld128(p0); kf.q[1] = ld128(p0 + 8);            // d 0..31
        v8f s = (nt == 0) ? s0 : s1;
        s = __builtin_amdgcn_wmma_f32_16x16x32_bf16(false, qf[0].bf, false, kf.bf, (short)0, s, false, false);
        const unsigned short* p1 = &sm.Ks[cur][nt * 16 + lm][32 + lh * 16];
        kf.q[0] = ld128(p1); kf.q[1] = ld128(p1 + 8);            // d 32..63
        s = __builtin_amdgcn_wmma_f32_16x16x32_bf16(false, qf[1].bf, false, kf.bf, (short)0, s, false, false);
        if (nt == 0) s0 = s; else s1 = s;
      }

      const float scale = 0.125f;        // 1/sqrt(64)
#pragma unroll
      for (int i = 0; i < 8; ++i) {
        int row = qBase + i + 8 * lh;
        int c0 = ktBase + lm, c1 = ktBase + 16 + lm;
        float x0 = s0[i] * scale; if (c0 > row) x0 = NEG_INF;
        float x1 = s1[i] * scale; if (c1 > row) x1 = NEG_INF;
        float mx = fmaxf(x0, x1);
#pragma unroll
        for (int off = 1; off < 16; off <<= 1) mx = fmaxf(mx, __shfl_xor(mx, off, 16));
        float mr = fmaxf(mstat[i], mx);
        float msafe = (mr == NEG_INF) ? 0.0f : mr;
        float alpha = __expf(mstat[i] - msafe);
        float e0 = __expf(x0 - msafe);
        float e1 = __expf(x1 - msafe);
        float sm2 = e0 + e1;
#pragma unroll
        for (int off = 1; off < 16; off <<= 1) sm2 += __shfl_xor(sm2, off, 16);
        lstat[i] = lstat[i] * alpha + sm2;
        mstat[i] = mr;
        oacc[0][i] *= alpha; oacc[1][i] *= alpha;
        oacc[2][i] *= alpha; oacc[3][i] *= alpha;
        sm.Pb[wave][i + 8 * lh][lm]      = f2bf(e0);  // C-layout -> A-layout via LDS
        sm.Pb[wave][i + 8 * lh][16 + lm] = f2bf(e1);
      }

      FragU pf;
      const unsigned short* pp = &sm.Pb[wave][lm][lh * 16];
      pf.q[0] = ld128(pp); pf.q[1] = ld128(pp + 8);

#pragma unroll
      for (int c = 0; c < 4; ++c) {      // O[:, c*16..+15] += P * V
        FragU vf;
        const unsigned short* vp = &sm.Vt[cur][c * 16 + lm][lh * 16];
        vf.q[0] = ld128(vp); vf.q[1] = ld128(vp + 8);
        oacc[c] = __builtin_amdgcn_wmma_f32_16x16x32_bf16(
            false, pf.bf, false, vf.bf, (short)0, oacc[c], false, false);
      }
    }
  }

  // normalize + store into [B, T, C] (C-col = h*64+d) as bf16 for proj GEMM
  const int b = bh >> 4, h = bh & 15;
#pragma unroll
  for (int i = 0; i < 8; ++i) {
    float inv = 1.0f / lstat[i];
    int t = qBase + i + 8 * lh;
    size_t base = ((size_t)b * SEQ + t) * EMBED + h * HDIM;
#pragma unroll
    for (int c = 0; c < 4; ++c)
      O[base + c * 16 + lm] = f2bf(oacc[c][i] * inv);
  }
}

// ---------------- host launch ----------------
extern "C" void kernel_launch(void* const* d_in, const int* in_sizes, int n_in,
                              void* d_out, int out_size, void* d_ws, size_t ws_size,
                              hipStream_t stream) {
  const float* x     = (const float*)d_in[0];   // [4,2048,1024]
  const float* Wqkv  = (const float*)d_in[1];   // [3072,1024]
  const float* Wproj = (const float*)d_in[2];   // [1024,1024]
  float* out = (float*)d_out;                   // [4,2048,1024] f32

  const size_t M  = (size_t)BATCH * SEQ;        // 8192
  const size_t NX = M * EMBED;                  // 8388608
  const size_t NW = (size_t)3 * EMBED * EMBED;  // 3145728
  const size_t NP = (size_t)EMBED * EMBED;      // 1048576

  unsigned short* xb     = (unsigned short*)d_ws;
  unsigned short* wqkvb  = xb + NX;
  unsigned short* wprojb = wqkvb + NW;
  unsigned short* qb     = wprojb + NP;
  unsigned short* kb     = qb + NX;
  unsigned short* vb     = kb + NX;
  unsigned short* attnb  = vb + NX;

  cvt_bf16<<<dim3((unsigned)(NX / 4 / 256)), 256, 0, stream>>>(x, xb, (int)(NX / 4));
  cvt_bf16<<<dim3((unsigned)(NW / 4 / 256)), 256, 0, stream>>>(Wqkv, wqkvb, (int)(NW / 4));
  cvt_bf16<<<dim3((unsigned)(NP / 4 / 256)), 256, 0, stream>>>(Wproj, wprojb, (int)(NP / 4));

  // QKV projection: [8192,1024] x [3072,1024]^T -> split q/k/v [B,H,T,D] bf16
  gemm_wmma<<<dim3(64, 12), 256, 0, stream>>>(xb, wqkvb, 3 * EMBED, 1,
                                              nullptr, qb, kb, vb);
  // Fused causal flash attention -> attnb [B,T,C] bf16
  flash_attn<<<dim3(SEQ / 128, BATCH * NHEAD), 256, 0, stream>>>(qb, kb, vb, attnb);
  // Output projection: [8192,1024] x [1024,1024]^T -> f32 out
  gemm_wmma<<<dim3(64, 4), 256, 0, stream>>>(attnb, wprojb, EMBED, 0,
                                             out, nullptr, nullptr, nullptr);
}
